// Curvphormer_90623809946326
// MI455X (gfx1250) — compile-verified, matrix operand order checked by reading, not run on hardware
//
#include <hip/hip_runtime.h>
#include <hip/hip_bf16.h>
#include <stdint.h>

// ---------------------------------------------------------------------------
// Curvphormer on MI455X (gfx1250): f16 WMMA for all dense GEMMs, async-to-LDS
// weight staging (ASYNCcnt path), fp32 edge softmax/scatter phase with
// L2-resident gathers, algebraically-folded curvature bias.
// All gemm outputs padded to 128-row multiples -> branchless store epilogue.
// ---------------------------------------------------------------------------

typedef _Float16 half_t;
typedef __attribute__((ext_vector_type(16))) _Float16 v16h;
typedef __attribute__((ext_vector_type(8)))  float    v8f;

#define DMODEL 64
#define NHEAD  4
#define HEADD  16

union U32B { uint4 u[2]; v16h v; };
union U16B { uint4 u; half_t h[8]; };

// CDNA5 16-bit A-frag (ISA 7.12.2): lane l holds row m=l&15; K offsets are two
// contiguous 8-half (16B) runs [kb..kb+7],[16+kb..16+kb+7], kb=(l&16)?8:0.
__device__ inline v16h load_frag_f16(const half_t* rowp, int kbase, int kb) {
    U32B t;
    t.u[0] = *(const uint4*)(rowp + kbase + kb);
    t.u[1] = *(const uint4*)(rowp + kbase + 16 + kb);
    return t.v;
}

__device__ inline v16h load_frag_f32(const float* rowp, int kbase, int kb) {
    v16h r;
    const float4* p0 = (const float4*)(rowp + kbase + kb);
    const float4* p1 = (const float4*)(rowp + kbase + 16 + kb);
    float4 a0 = p0[0], a1 = p0[1], b0 = p1[0], b1 = p1[1];
    r[0] = (_Float16)a0.x; r[1] = (_Float16)a0.y; r[2] = (_Float16)a0.z; r[3] = (_Float16)a0.w;
    r[4] = (_Float16)a1.x; r[5] = (_Float16)a1.y; r[6] = (_Float16)a1.z; r[7] = (_Float16)a1.w;
    r[8]  = (_Float16)b0.x; r[9]  = (_Float16)b0.y; r[10] = (_Float16)b0.z; r[11] = (_Float16)b0.w;
    r[12] = (_Float16)b1.x; r[13] = (_Float16)b1.y; r[14] = (_Float16)b1.z; r[15] = (_Float16)b1.w;
    return r;
}

// async DMA: one 16B chunk global -> LDS, tracked by ASYNCcnt (CDNA5)
__device__ inline void async_copy_b128(unsigned lds_off, const void* gaddr) {
    asm volatile("global_load_async_to_lds_b128 %0, %1, off"
                 :: "v"(lds_off), "v"(gaddr) : "memory");
}
__device__ inline void wait_async0() {
    asm volatile("s_wait_asynccnt 0x0" ::: "memory");
}

// ---------------------------------------------------------------------------
// Generic WMMA GEMM:  out[.. x NOUT] = act(A[.. x K] @ W[K x NOUT] + b) (+res)
// Block = 256 thr = 8 waves; each wave owns 16 rows; out is padded to 128-row
// multiples so the epilogue is branchless with immediate-offset stores.
// W16cm: f16 column-major padded [NOUT][K+8] image, async-DMA'd into LDS.
// ---------------------------------------------------------------------------
template <int K, int NOUT, bool RELU, bool RESID, bool A_F16, bool O_F16>
__global__ __launch_bounds__(256) void gemm_wmma_k(
    const void* __restrict__ A, const half_t* __restrict__ W16cm,
    const float* __restrict__ bias, void* __restrict__ out,
    const float* __restrict__ resid, int nrows)
{
    constexpr int KP = K + 8;
    constexpr int WBYTES = NOUT * KP * 2;
    __shared__ __align__(16) half_t Wlds[NOUT * KP];

    const int tid = threadIdx.x;
    {
        const char* gsrc = (const char*)W16cm;
        unsigned lds_base = (unsigned)(uintptr_t)(&Wlds[0]);   // LDS aperture: low 32 bits
        for (int o = tid * 16; o < WBYTES; o += 256 * 16)
            async_copy_b128(lds_base + o, gsrc + o);
        wait_async0();
    }
    __syncthreads();

    const int wave = tid >> 5, lane = tid & 31;
    const int row0 = blockIdx.x * 128 + wave * 16;
    const int m_a  = lane & 15;
    const int kb   = (lane & 16) ? 8 : 0;
    const int rowA = min(row0 + m_a, nrows - 1);  // clamp: EXEC stays uniform

    const half_t* arow16 = A_F16 ? ((const half_t*)A) + (size_t)rowA * K : nullptr;
    const float*  arow32 = A_F16 ? nullptr : ((const float*)A) + (size_t)rowA * K;

    // A fragments depend only on ks -> load once
    v16h afr[K / 32];
#pragma unroll
    for (int ks = 0; ks < K / 32; ++ks) {
        if (A_F16) afr[ks] = load_frag_f16(arow16, ks * 32, kb);
        else       afr[ks] = load_frag_f32(arow32, ks * 32, kb);
    }

    const int n_lane = lane & 15;
    const int rbase  = row0 + ((lane & 16) ? 8 : 0);   // first of this lane's 8 C rows

#pragma unroll
    for (int nt = 0; nt < NOUT / 16; ++nt) {
        v8f acc;
#pragma unroll
        for (int r = 0; r < 8; ++r) acc[r] = 0.0f;

        const half_t* wcol = Wlds + (nt * 16 + n_lane) * KP;

#pragma unroll
        for (int ks = 0; ks < K / 32; ++ks) {
            v16h b = load_frag_f16(wcol, ks * 32, kb);
            acc = __builtin_amdgcn_wmma_f32_16x16x32_f16(
                false, afr[ks], false, b, (short)0, acc, false, false);
        }

        const int col = nt * 16 + n_lane;
        const float bcol = bias[col];
        const size_t base = (size_t)rbase * NOUT + col;
        const float* rp = RESID ? (resid + base) : nullptr;
        half_t* oph = O_F16 ? ((half_t*)out + base) : nullptr;
        float*  opf = O_F16 ? nullptr : ((float*)out + base);
#pragma unroll
        for (int r = 0; r < 8; ++r) {
            float val = acc[r] + bcol;
            if (RELU)  val = fmaxf(val, 0.0f);
            if (RESID) val += rp[r * NOUT];
            if (O_F16) oph[r * NOUT] = (half_t)val;
            else       opf[r * NOUT] = val;
        }
    }
}

// ---------------------------------------------------------------------------
// Small support kernels
// ---------------------------------------------------------------------------
__global__ void f32_to_f16_k(const float* __restrict__ s, half_t* __restrict__ d, int n) {
    int i = blockIdx.x * blockDim.x + threadIdx.x;
    if (i < n) d[i] = (half_t)s[i];
}

// weight fp32 row-major [K][NOUT] -> f16 column-major padded [NOUT][K+8]
template <int K, int NOUT>
__global__ void w_colmajor_f16_k(const float* __restrict__ s, half_t* __restrict__ d) {
    constexpr int KP = K + 8;
    int idx = blockIdx.x * blockDim.x + threadIdx.x;
    if (idx >= NOUT * KP) return;
    int n = idx / KP, k = idx - n * KP;
    d[idx] = (half_t)((k < K) ? s[k * NOUT + n] : 0.0f);
}

// LayerNorm over 64 cols, f16 output
__global__ void ln_k(const float* __restrict__ h, const float* __restrict__ g,
                     const float* __restrict__ b, half_t* __restrict__ xn, int n) {
    int row = blockIdx.x * blockDim.x + threadIdx.x;
    if (row >= n) return;
    const float4* rp = (const float4*)(h + (size_t)row * DMODEL);
    float s = 0.f, s2 = 0.f;
#pragma unroll
    for (int i = 0; i < 16; ++i) {
        float4 v = rp[i];
        s  += v.x + v.y + v.z + v.w;
        s2 += v.x * v.x + v.y * v.y + v.z * v.z + v.w * v.w;
    }
    float mu = s * (1.0f / DMODEL);
    float var = s2 * (1.0f / DMODEL) - mu * mu;
    float rstd = rsqrtf(var + 1e-5f);
    half_t* op = xn + (size_t)row * DMODEL;
#pragma unroll
    for (int i = 0; i < 16; ++i) {
        float4 v = rp[i];
        op[4 * i + 0] = (half_t)((v.x - mu) * rstd * g[4 * i + 0] + b[4 * i + 0]);
        op[4 * i + 1] = (half_t)((v.y - mu) * rstd * g[4 * i + 1] + b[4 * i + 1]);
        op[4 * i + 2] = (half_t)((v.z - mu) * rstd * g[4 * i + 2] + b[4 * i + 2]);
        op[4 * i + 3] = (half_t)((v.w - mu) * rstd * g[4 * i + 3] + b[4 * i + 3]);
    }
}

// Fold curvature bias:  Mb[l] = curv_W2 @ bias_W_l (64x4), Cb[l] = curv_b2@bias_W_l + bias_b_l
__global__ void bias_fold_k(const float* __restrict__ W2, const float* __restrict__ b2,
                            const float* __restrict__ bW0, const float* __restrict__ bb0,
                            const float* __restrict__ bW1, const float* __restrict__ bb1,
                            float* __restrict__ Mb, float* __restrict__ Cb) {
    int idx = threadIdx.x;
    for (int t = idx; t < 2 * DMODEL * NHEAD; t += blockDim.x) {
        int l = t >> 8, rem = t & 255, d = rem >> 2, hh = rem & 3;
        const float* bW = l ? bW1 : bW0;
        float acc = 0.f;
        for (int j = 0; j < DMODEL; ++j) acc += W2[d * DMODEL + j] * bW[j * NHEAD + hh];
        Mb[t] = acc;
    }
    if (idx < 8) {
        int l = idx >> 2, hh = idx & 3;
        const float* bW = l ? bW1 : bW0;
        const float* bb = l ? bb1 : bb0;
        float acc = bb[hh];
        for (int j = 0; j < DMODEL; ++j) acc += b2[j] * bW[j * NHEAD + hh];
        Cb[idx] = acc;
    }
}

// Per-edge curvature score bias for both layers: sbias[l][e][h]
__global__ void edge_bias_k(const float* __restrict__ curv, const float* __restrict__ W1,
                            const float* __restrict__ b1, const float* __restrict__ Mb,
                            const float* __restrict__ Cb, float* __restrict__ sbias, int E) {
    __shared__ float sW1[DMODEL], sb1[DMODEL], sMb[2 * DMODEL * NHEAD], sCb[8];
    int tid = threadIdx.x;
    if (tid < DMODEL) { sW1[tid] = W1[tid]; sb1[tid] = b1[tid]; }
    for (int t = tid; t < 2 * DMODEL * NHEAD; t += blockDim.x) sMb[t] = Mb[t];
    if (tid < 8) sCb[tid] = Cb[tid];
    __syncthreads();
    int e = blockIdx.x * blockDim.x + tid;
    if (e >= E) return;
    float c = curv[e];
    float a0[NHEAD], a1[NHEAD];
#pragma unroll
    for (int h = 0; h < NHEAD; ++h) { a0[h] = sCb[h]; a1[h] = sCb[4 + h]; }
    for (int d = 0; d < DMODEL; ++d) {
        float r = fmaxf(c * sW1[d] + sb1[d], 0.0f);
#pragma unroll
        for (int h = 0; h < NHEAD; ++h) {
            a0[h] += r * sMb[d * NHEAD + h];
            a1[h] += r * sMb[(DMODEL + d) * NHEAD + h];
        }
    }
#pragma unroll
    for (int h = 0; h < NHEAD; ++h) {
        sbias[(size_t)e * NHEAD + h] = a0[h];
        sbias[(size_t)E * NHEAD + (size_t)e * NHEAD + h] = a1[h];
    }
}

// order-preserving float->uint key for atomic max
__device__ inline unsigned fkey(float f) {
    unsigned b = __float_as_uint(f);
    return (b & 0x80000000u) ? ~b : (b | 0x80000000u);
}
__device__ inline float funkey(unsigned k) {
    unsigned b = (k & 0x80000000u) ? (k & 0x7FFFFFFFu) : ~k;
    return __uint_as_float(b);
}

// pass 1: per-edge per-head scores (+ folded curvature bias), segment max by tgt
__global__ void edge_scores_k(const int* __restrict__ src, const int* __restrict__ tgt,
                              const half_t* __restrict__ q16, const half_t* __restrict__ k16,
                              const float* __restrict__ sbias_l, float* __restrict__ scores,
                              unsigned* __restrict__ mxbits, int E) {
    int e = blockIdx.x * blockDim.x + threadIdx.x;
    if (e >= E) return;
    int s = src[e], t = tgt[e];
    const uint4* qp = (const uint4*)(q16 + (size_t)s * DMODEL);
    const uint4* kp = (const uint4*)(k16 + (size_t)t * DMODEL);
    float acc[NHEAD] = {0.f, 0.f, 0.f, 0.f};
#pragma unroll
    for (int c = 0; c < 8; ++c) {
        U16B a, b; a.u = qp[c]; b.u = kp[c];
        int h = c >> 1;
#pragma unroll
        for (int j = 0; j < 8; ++j) acc[h] += (float)a.h[j] * (float)b.h[j];
    }
#pragma unroll
    for (int h = 0; h < NHEAD; ++h) {
        float sv = acc[h] * 0.25f + sbias_l[(size_t)e * NHEAD + h];
        scores[(size_t)e * NHEAD + h] = sv;
        atomicMax(&mxbits[(size_t)t * NHEAD + h], fkey(sv));
    }
}

// pass 2: ex = exp(s - mx[tgt]); den[tgt] += ex  (scores buffer overwritten with ex)
__global__ void edge_exp_k(const int* __restrict__ tgt, float* __restrict__ scores,
                           const unsigned* __restrict__ mxbits, float* __restrict__ den, int E) {
    int e = blockIdx.x * blockDim.x + threadIdx.x;
    if (e >= E) return;
    int t = tgt[e];
#pragma unroll
    for (int h = 0; h < NHEAD; ++h) {
        float mx = funkey(mxbits[(size_t)t * NHEAD + h]);
        float ex = __expf(scores[(size_t)e * NHEAD + h] - mx);
        scores[(size_t)e * NHEAD + h] = ex;
        atomicAdd(&den[(size_t)t * NHEAD + h], ex);
    }
}

// pass 3: prob = ex/den[tgt]; agg[src] += prob (x) v[tgt]
__global__ void edge_msg_k(const int* __restrict__ src, const int* __restrict__ tgt,
                           const float* __restrict__ ex, const float* __restrict__ den,
                           const half_t* __restrict__ v16, float* __restrict__ agg, int E) {
    int e = blockIdx.x * blockDim.x + threadIdx.x;
    if (e >= E) return;
    int s = src[e], t = tgt[e];
    float p[NHEAD];
#pragma unroll
    for (int h = 0; h < NHEAD; ++h)
        p[h] = ex[(size_t)e * NHEAD + h] / den[(size_t)t * NHEAD + h];
    const uint4* vp = (const uint4*)(v16 + (size_t)t * DMODEL);
    float* ap = agg + (size_t)s * DMODEL;
#pragma unroll
    for (int c = 0; c < 8; ++c) {
        U16B vv; vv.u = vp[c];
        int h = c >> 1;
#pragma unroll
        for (int j = 0; j < 8; ++j) atomicAdd(&ap[c * 8 + j], p[h] * (float)vv.h[j]);
    }
}

// final tiny head: out[i] = hid[i,:32] @ out_W2 + out_b2
__global__ void head2_k(const half_t* __restrict__ hid, const float* __restrict__ W2,
                        const float* __restrict__ b2, float* __restrict__ out, int n) {
    int i = blockIdx.x * blockDim.x + threadIdx.x;
    if (i >= n) return;
    float acc = b2[0];
    const half_t* hp = hid + (size_t)i * 32;
#pragma unroll
    for (int j = 0; j < 32; ++j) acc += (float)hp[j] * W2[j];
    out[i] = acc;
}

// ---------------------------------------------------------------------------
// host side
// ---------------------------------------------------------------------------
extern "C" void kernel_launch(void* const* d_in, const int* in_sizes, int n_in,
                              void* d_out, int out_size, void* d_ws, size_t ws_size,
                              hipStream_t stream) {
    (void)n_in; (void)ws_size;
    const int N = in_sizes[0] / 128;
    const int E = in_sizes[2];
    const int Npad = ((N + 127) / 128) * 128;   // gemm tail blocks store into padding

    const float* x    = (const float*)d_in[0];
    const int*   src  = (const int*)d_in[1];
    const int*   tgt  = src + E;
    const float* curv = (const float*)d_in[2];
    const float* node_W = (const float*)d_in[3];
    const float* node_b = (const float*)d_in[4];
    const float* curv_W1 = (const float*)d_in[5];
    const float* curv_b1 = (const float*)d_in[6];
    const float* curv_W2 = (const float*)d_in[7];
    const float* curv_b2 = (const float*)d_in[8];
    auto LP = [&](int l, int j) -> const float* { return (const float*)d_in[9 + 18 * l + j]; };
    const float* out_W1 = (const float*)d_in[45];
    const float* out_b1 = (const float*)d_in[46];
    const float* out_W2 = (const float*)d_in[47];
    const float* out_b2 = (const float*)d_in[48];

    // --- workspace bump allocator ---
    char* wsb = (char*)d_ws; size_t off = 0;
    auto alloc = [&](size_t bytes) -> void* {
        void* p = wsb + off; off = (off + bytes + 255) & ~(size_t)255; return p;
    };
    half_t* x16   = (half_t*)alloc((size_t)N * 128 * 2);
    float*  h     = (float*)alloc((size_t)Npad * DMODEL * 4);
    half_t* xn16  = (half_t*)alloc((size_t)Npad * DMODEL * 2);
    half_t* q16   = (half_t*)alloc((size_t)Npad * DMODEL * 2);
    half_t* k16   = (half_t*)alloc((size_t)Npad * DMODEL * 2);
    half_t* v16   = (half_t*)alloc((size_t)Npad * DMODEL * 2);
    half_t* mid16 = (half_t*)alloc((size_t)Npad * 256 * 2);
    half_t* hid16 = (half_t*)alloc((size_t)Npad * 32 * 2);
    float*  sbias = (float*)alloc((size_t)2 * E * NHEAD * 4);
    float*  scores = (float*)alloc((size_t)E * NHEAD * 4);
    unsigned* mxbits = (unsigned*)alloc((size_t)N * NHEAD * 4);
    float*  den   = (float*)alloc((size_t)N * NHEAD * 4);
    float*  agg   = (float*)alloc((size_t)N * DMODEL * 4);
    float*  Mb    = (float*)alloc(2 * DMODEL * NHEAD * 4);
    float*  Cb    = (float*)alloc(8 * 4);
    // f16 weights, column-major padded [NOUT][K+8]
    half_t* nodeW16 = (half_t*)alloc((size_t)64 * (128 + 8) * 2);
    half_t* qkvoW16[2][4];
    half_t* f1W16[2]; half_t* f2W16[2];
    for (int l = 0; l < 2; ++l) {
        for (int j = 0; j < 4; ++j) qkvoW16[l][j] = (half_t*)alloc((size_t)64 * (64 + 8) * 2);
        f1W16[l] = (half_t*)alloc((size_t)256 * (64 + 8) * 2);
        f2W16[l] = (half_t*)alloc((size_t)64 * (256 + 8) * 2);
    }
    half_t* outW1_16 = (half_t*)alloc((size_t)32 * (64 + 8) * 2);

    // activations conversion (row-major)
    f32_to_f16_k<<<(N * 128 + 255) / 256, 256, 0, stream>>>(x, x16, N * 128);

    // weight conversions to col-major padded f16
    auto wcm = [&](auto kern, const float* s, half_t* d, int total) {
        kern<<<(total + 255) / 256, 256, 0, stream>>>(s, d);
    };
    wcm(w_colmajor_f16_k<128, 64>, node_W, nodeW16, 64 * 136);
    for (int l = 0; l < 2; ++l) {
        wcm(w_colmajor_f16_k<64, 64>, LP(l, 0), qkvoW16[l][0], 64 * 72);
        wcm(w_colmajor_f16_k<64, 64>, LP(l, 2), qkvoW16[l][1], 64 * 72);
        wcm(w_colmajor_f16_k<64, 64>, LP(l, 4), qkvoW16[l][2], 64 * 72);
        wcm(w_colmajor_f16_k<64, 64>, LP(l, 6), qkvoW16[l][3], 64 * 72);
        wcm(w_colmajor_f16_k<64, 256>, LP(l, 10), f1W16[l], 256 * 72);
        wcm(w_colmajor_f16_k<256, 64>, LP(l, 12), f2W16[l], 64 * 264);
    }
    wcm(w_colmajor_f16_k<64, 32>, out_W1, outW1_16, 32 * 72);

    // folded curvature bias
    bias_fold_k<<<1, 256, 0, stream>>>(curv_W2, curv_b2, LP(0, 8), LP(0, 9),
                                       LP(1, 8), LP(1, 9), Mb, Cb);
    edge_bias_k<<<(E + 255) / 256, 256, 0, stream>>>(curv, curv_W1, curv_b1, Mb, Cb, sbias, E);

    const int gN = (N + 127) / 128;
    const int gE = (E + 255) / 256;
    const int gNt = (N + 255) / 256;

    // node projection: h = x @ node_W + node_b
    gemm_wmma_k<128, 64, false, false, true, false>
        <<<gN, 256, 0, stream>>>(x16, nodeW16, node_b, h, nullptr, N);

    for (int l = 0; l < 2; ++l) {
        // LN1 -> xn16
        ln_k<<<gNt, 256, 0, stream>>>(h, LP(l, 14), LP(l, 15), xn16, N);
        // q/k/v projections -> f16
        gemm_wmma_k<64, 64, false, false, true, true>
            <<<gN, 256, 0, stream>>>(xn16, qkvoW16[l][0], LP(l, 1), q16, nullptr, N);
        gemm_wmma_k<64, 64, false, false, true, true>
            <<<gN, 256, 0, stream>>>(xn16, qkvoW16[l][1], LP(l, 3), k16, nullptr, N);
        gemm_wmma_k<64, 64, false, false, true, true>
            <<<gN, 256, 0, stream>>>(xn16, qkvoW16[l][2], LP(l, 5), v16, nullptr, N);
        // segment softmax state
        hipMemsetAsync(mxbits, 0, (size_t)N * NHEAD * 4, stream);
        hipMemsetAsync(den, 0, (size_t)N * NHEAD * 4, stream);
        hipMemsetAsync(agg, 0, (size_t)N * DMODEL * 4, stream);
        const float* sb_l = sbias + (size_t)l * E * NHEAD;
        edge_scores_k<<<gE, 256, 0, stream>>>(src, tgt, q16, k16, sb_l, scores, mxbits, E);
        edge_exp_k<<<gE, 256, 0, stream>>>(tgt, scores, mxbits, den, E);
        edge_msg_k<<<gE, 256, 0, stream>>>(src, tgt, scores, den, v16, agg, E);
        // h += agg @ o_W + o_b
        gemm_wmma_k<64, 64, false, true, false, false>
            <<<gN, 256, 0, stream>>>(agg, qkvoW16[l][3], LP(l, 7), h, h, N);
        // LN2 -> xn16 ; FFN
        ln_k<<<gNt, 256, 0, stream>>>(h, LP(l, 16), LP(l, 17), xn16, N);
        gemm_wmma_k<64, 256, true, false, true, true>
            <<<gN, 256, 0, stream>>>(xn16, f1W16[l], LP(l, 11), mid16, nullptr, N);
        gemm_wmma_k<256, 64, false, true, true, false>
            <<<gN, 256, 0, stream>>>(mid16, f2W16[l], LP(l, 13), h, h, N);
    }

    // output head
    gemm_wmma_k<64, 32, true, false, false, true>
        <<<gN, 256, 0, stream>>>(h, outW1_16, out_b1, hid16, nullptr, N);
    head2_k<<<(out_size + 255) / 256, 256, 0, stream>>>(hid16, out_W2, out_b2,
                                                        (float*)d_out, out_size);
}
